// GPT3CoreAttention_55327768708189
// MI455X (gfx1250) — matrix operand order repaired and checked by reading, not compiled, not run on hardware
//
#include <hip/hip_runtime.h>

typedef __bf16 bf16_t;
typedef __attribute__((ext_vector_type(16))) __bf16 v16bf;
typedef __attribute__((ext_vector_type(8)))  __bf16 v8bf;
typedef __attribute__((ext_vector_type(8)))  float  v8f;
typedef __attribute__((ext_vector_type(4)))  unsigned int v4u;
typedef __attribute__((ext_vector_type(8)))  int v8i;
typedef __attribute__((ext_vector_type(4)))  int v4i;

#define HEADS    16
#define DHEAD    64
#define QTILE    64
#define KTILE    64
#define NTHREADS 128

// padded LDS row strides (elements); rows stay 16B-aligned (stride*2 = 144B)
#define QSTR 72
#define KSTR 72
#define VSTR 72
#define PSTR 72

// LDS layout for the TDM kernel (element offsets into one shared array)
#define K0OFF 0
#define V0OFF (KTILE * KSTR)
#define K1OFF (2 * KTILE * KSTR)
#define V1OFF (3 * KTILE * KSTR)
#define PSOFF (4 * KTILE * KSTR)
#define SMELEMS (PSOFF + 4 * 16 * PSTR)

// ---- DPP16 16-lane reductions (stay within lane halves 0-15 / 16-31) ----
template <int CTRL>
__device__ inline float dpp_mov(float x) {
    int r = __builtin_amdgcn_update_dpp(0, __builtin_bit_cast(int, x),
                                        CTRL, 0xF, 0xF, true);
    return __builtin_bit_cast(float, r);
}
__device__ inline float red16_max(float x) {
    x = fmaxf(x, dpp_mov<0xB1>(x));    // quad_perm(1,0,3,2)
    x = fmaxf(x, dpp_mov<0x4E>(x));    // quad_perm(2,3,0,1)
    x = fmaxf(x, dpp_mov<0x141>(x));   // row_half_mirror
    x = fmaxf(x, dpp_mov<0x140>(x));   // row_mirror
    return x;
}
__device__ inline float red16_sum(float x) {
    x += dpp_mov<0xB1>(x);
    x += dpp_mov<0x4E>(x);
    x += dpp_mov<0x141>(x);
    x += dpp_mov<0x140>(x);
    return x;
}

// Build a v16bf A/B operand from two 16-byte chunks (LDS or global bf16)
__device__ inline v16bf pack2(const bf16_t* p, int off0, int off1) {
    const v8bf x = *(const v8bf*)(p + off0);
    const v8bf y = *(const v8bf*)(p + off1);
    v16bf r;
#pragma unroll
    for (int i = 0; i < 8; ++i) { r[i] = x[i]; r[i + 8] = y[i]; }
    return r;
}
__device__ inline v16bf pack16(const bf16_t* p, int off) {
    return pack2(p, off, off + 8);
}

// ---- Tensor Data Mover: 2D tile (64 x 64 x 2B) global -> LDS, padded rows ----
// pad: +4 DWORDs every 32 DWORDs => LDS row stride 72 elements (144B)
__device__ inline void tdm_load_2d(unsigned int lds_byte_off, const bf16_t* gptr,
                                   unsigned int tensor_d0, unsigned int tensor_d1,
                                   unsigned int d0_stride) {
    unsigned long long ga = (unsigned long long)gptr;
    v4u g0;
    g0[0] = 1u;                                    // count=1, user descriptor
    g0[1] = lds_byte_off;                          // lds_addr
    g0[2] = (unsigned int)(ga & 0xFFFFFFFFu);      // global_addr[31:0]
    g0[3] = (unsigned int)((ga >> 32) & 0x1FFFFFFu) | (2u << 30);  // addr[56:32] | type=2
    v8i g1;
    g1[0] = (int)((1u << 16) | (1u << 20) | (4u << 22) | (3u << 25));
            // data_size=2B | pad_enable | pad_interval=32dw | pad_amount=4dw
    g1[1] = (int)((tensor_d0 & 0xFFFFu) << 16);                    // tensor_dim0 lo16
    g1[2] = (int)(((tensor_d0 >> 16) & 0xFFFFu) |
                  ((tensor_d1 & 0xFFFFu) << 16));                  // dim0 hi | dim1 lo
    g1[3] = (int)(((tensor_d1 >> 16) & 0xFFFFu) | (64u << 16));    // dim1 hi | tile_dim0=64
    g1[4] = (int)64;                                               // tile_dim1=64, tile_dim2=0
    g1[5] = (int)d0_stride;                                        // dim0_stride[31:0]
    g1[6] = 0;                                                     // stride hi | dim1_stride lo
    g1[7] = 0;
    v4i gz4 = {0, 0, 0, 0};
    v8i gz8 = {0, 0, 0, 0, 0, 0, 0, 0};
    __builtin_amdgcn_tensor_load_to_lds(g0, g1, gz4, gz4, gz8, 0);
}

// =====================================================================
// Pre-pass: fp32 [s][b][h][d] -> bf16 head-major Qw/Kw [bh][s][d], Vw [bh][d][s]
// =====================================================================
__global__ __launch_bounds__(256)
void convert_pack_kernel(const float* __restrict__ Q, const float* __restrict__ K,
                         const float* __restrict__ V,
                         bf16_t* __restrict__ Qw, bf16_t* __restrict__ Kw,
                         bf16_t* __restrict__ Vw, int sq, int B) {
    long e = ((long)blockIdx.x * 256 + threadIdx.x) * 4;
    long n = (long)sq * B * HEADS * DHEAD;
    if (e >= n) return;
    int dch = (int)(e & 63);
    long r  = e >> 6;                  // (s*B + bb)*HEADS + hh
    int hh  = (int)(r % HEADS);
    long r2 = r / HEADS;
    int bb  = (int)(r2 % B);
    int s   = (int)(r2 / B);
    int bh  = bb * HEADS + hh;
    float4 q4 = *(const float4*)(Q + e);
    float4 k4 = *(const float4*)(K + e);
    float4 v4 = *(const float4*)(V + e);
    long o = ((long)bh * sq + s) * DHEAD + dch;
    Qw[o + 0] = (bf16_t)q4.x; Qw[o + 1] = (bf16_t)q4.y;
    Qw[o + 2] = (bf16_t)q4.z; Qw[o + 3] = (bf16_t)q4.w;
    Kw[o + 0] = (bf16_t)k4.x; Kw[o + 1] = (bf16_t)k4.y;
    Kw[o + 2] = (bf16_t)k4.z; Kw[o + 3] = (bf16_t)k4.w;
    long vo = ((long)bh * DHEAD + dch) * sq + s;
    Vw[vo + 0L * sq] = (bf16_t)v4.x;
    Vw[vo + 1L * sq] = (bf16_t)v4.y;
    Vw[vo + 2L * sq] = (bf16_t)v4.z;
    Vw[vo + 3L * sq] = (bf16_t)v4.w;
}

// =====================================================================
// Main kernel, TDM path: double-buffered tensor_load_to_lds + WMMA
// =====================================================================
__global__ __launch_bounds__(NTHREADS)
void attn_tdm_kernel(const bf16_t* __restrict__ Qw, const bf16_t* __restrict__ Kw,
                     const bf16_t* __restrict__ Vw, float* __restrict__ Out,
                     int sq, int B) {
    __shared__ bf16_t SM[SMELEMS];

    const int qb   = blockIdx.x * QTILE;
    const int bh   = blockIdx.y;
    const int bb   = bh / HEADS;
    const int hh   = bh % HEADS;
    const int tid  = threadIdx.x;
    const int wave = tid >> 5;
    const int lane = tid & 31;
    const int half = lane >> 4;
    const int l16  = lane & 15;

    // Q A-operands straight from global bf16 (head-major rows are contiguous)
    const int qrow = wave * 16 + l16;
    int srow = qb + qrow; if (srow > sq - 1) srow = sq - 1;
    const bf16_t* Qrow = Qw + ((long)bh * sq + srow) * DHEAD;
    const v16bf qa0 = pack16(Qrow,      half * 16);
    const v16bf qa1 = pack16(Qrow, 32 + half * 16);

    v8f o[4] = {{}, {}, {}, {}};
    float rmax[8], rsum[8];
#pragma unroll
    for (int v = 0; v < 8; ++v) { rmax[v] = -3.0e38f; rsum[v] = 0.f; }

    int limit = qb + QTILE; if (limit > sq) limit = sq;
    const int ntiles = (limit + KTILE - 1) / KTILE;

    const bf16_t* Khead = Kw + (long)bh * sq * DHEAD;
    const bf16_t* Vhead = Vw + (long)bh * DHEAD * sq;

    // issue tile 0 into buffer 0 (one TDM pair per workgroup, wave 0 only)
    if (wave == 0) {
        tdm_load_2d((unsigned int)(K0OFF * 2), Khead, DHEAD, (unsigned int)sq, DHEAD);
        tdm_load_2d((unsigned int)(V0OFF * 2), Vhead, (unsigned int)sq, DHEAD,
                    (unsigned int)sq);
    }

    for (int t = 0; t < ntiles; ++t) {
        const int kb  = t * KTILE;
        const int buf = t & 1;
        if (wave == 0) __builtin_amdgcn_s_wait_tensorcnt(0);
        __syncthreads();                       // tile t visible to all waves

        // overlap: DMA tile t+1 into the other buffer while we compute tile t
        if (wave == 0 && t + 1 < ntiles) {
            const int nkb = kb + KTILE;
            tdm_load_2d((unsigned int)((buf ? K0OFF : K1OFF) * 2),
                        Khead + (long)nkb * DHEAD, DHEAD,
                        (unsigned int)(sq - nkb), DHEAD);
            tdm_load_2d((unsigned int)((buf ? V0OFF : V1OFF) * 2),
                        Vhead + nkb, (unsigned int)(sq - nkb), DHEAD,
                        (unsigned int)sq);
        }

        const bf16_t* Kb = SM + (buf ? K1OFF : K0OFF);
        const bf16_t* Vb = SM + (buf ? V1OFF : V0OFF);

        // ---- S = Q * K^T : 4 key groups x (K-dim 64 = 2 WMMAs) ----
        v8f s[4];
#pragma unroll
        for (int g = 0; g < 4; ++g) {
            v16bf b0 = pack16(Kb, (g * 16 + l16) * KSTR +      half * 16);
            v16bf b1 = pack16(Kb, (g * 16 + l16) * KSTR + 32 + half * 16);
            v8f acc = {};
            acc = __builtin_amdgcn_wmma_f32_16x16x32_bf16(false, qa0, false, b0, (short)0, acc, false, false);
            acc = __builtin_amdgcn_wmma_f32_16x16x32_bf16(false, qa1, false, b1, (short)0, acc, false, false);
            s[g] = acc;
        }

        // ---- scale, causal mask (diagonal tile only), online softmax ----
        const float scale = 0.125f;            // 1/sqrt(64); layer coeff cancels
        const bool need_mask = (kb + KTILE) > qb;
        float p[4][8], tmax[8];
#pragma unroll
        for (int v = 0; v < 8; ++v) {
            int rg = qb + wave * 16 + v + 8 * half;
            float m = -3.0e38f;
#pragma unroll
            for (int g = 0; g < 4; ++g) {
                float x = s[g][v] * scale;
                if (need_mask && (kb + g * 16 + l16 > rg)) x = -3.0e38f;
                p[g][v] = x;
                m = fmaxf(m, x);
            }
            tmax[v] = m;
        }
#pragma unroll
        for (int v = 0; v < 8; ++v) tmax[v] = red16_max(tmax[v]);

        float alpha[8], lsum[8];
#pragma unroll
        for (int v = 0; v < 8; ++v) {
            float nm = fmaxf(rmax[v], tmax[v]);
            alpha[v] = __expf(rmax[v] - nm);
            rmax[v]  = nm;
            float acc = 0.f;
#pragma unroll
            for (int g = 0; g < 4; ++g) {
                float e = __expf(p[g][v] - nm);
                p[g][v] = e;
                acc += e;
            }
            lsum[v] = acc;
        }
#pragma unroll
        for (int v = 0; v < 8; ++v) lsum[v] = red16_sum(lsum[v]);
#pragma unroll
        for (int v = 0; v < 8; ++v) {
            rsum[v] = rsum[v] * alpha[v] + lsum[v];
            o[0][v] *= alpha[v]; o[1][v] *= alpha[v];
            o[2][v] *= alpha[v]; o[3][v] *= alpha[v];
        }

        // ---- transpose P (C-layout -> A-layout) through per-wave LDS ----
        bf16_t* Pw = SM + PSOFF + wave * 16 * PSTR;
#pragma unroll
        for (int v = 0; v < 8; ++v) {
            int pr = v + 8 * half;
#pragma unroll
            for (int g = 0; g < 4; ++g)
                Pw[pr * PSTR + g * 16 + l16] = (bf16_t)p[g][v];
        }
        __syncthreads();

        const int pbase = l16 * PSTR;
        v16bf pa0 = pack16(Pw, pbase +      half * 16);
        v16bf pa1 = pack16(Pw, pbase + 32 + half * 16);

        // ---- O += P * V : 4 output chunks x (K-dim 64 = 2 WMMAs) ----
#pragma unroll
        for (int j = 0; j < 4; ++j) {
            v16bf vb0 = pack16(Vb, (j * 16 + l16) * VSTR +      half * 16);
            v16bf vb1 = pack16(Vb, (j * 16 + l16) * VSTR + 32 + half * 16);
            o[j] = __builtin_amdgcn_wmma_f32_16x16x32_bf16(false, pa0, false, vb0, (short)0, o[j], false, false);
            o[j] = __builtin_amdgcn_wmma_f32_16x16x32_bf16(false, pa1, false, vb1, (short)0, o[j], false, false);
        }
    }

    // ---- normalize and store: out[s][bb][hh*64 + ch], fp32 ----
#pragma unroll
    for (int v = 0; v < 8; ++v) {
        int rg = qb + wave * 16 + v + 8 * half;
        if (rg < sq) {
            float inv = 1.0f / rsum[v];
            long off = ((long)rg * B + bb) * (HEADS * DHEAD) + hh * DHEAD + l16;
            Out[off +  0] = o[0][v] * inv;
            Out[off + 16] = o[1][v] * inv;
            Out[off + 32] = o[2][v] * inv;
            Out[off + 48] = o[3][v] * inv;
        }
    }
}

// =====================================================================
// Fallback (no workspace): self-contained kernel, converts fp32->bf16 inline
// =====================================================================
__global__ __launch_bounds__(NTHREADS)
void attn_fallback_kernel(const float* __restrict__ Q, const float* __restrict__ K,
                          const float* __restrict__ V, float* __restrict__ Out,
                          int sq, int B) {
    __shared__ bf16_t Qs[QTILE * QSTR];
    __shared__ bf16_t Ks[KTILE * KSTR];
    __shared__ bf16_t Vts[DHEAD * VSTR];
    __shared__ bf16_t Ps[4 * 16 * PSTR];

    const int qb   = blockIdx.x * QTILE;
    const int bh   = blockIdx.y;
    const int bb   = bh / HEADS;
    const int hh   = bh % HEADS;
    const int tid  = threadIdx.x;
    const int wave = tid >> 5;
    const int lane = tid & 31;
    const int half = lane >> 4;
    const int l16  = lane & 15;

    const long rowStride = (long)B * HEADS * DHEAD;
    const float* Qg = Q + ((long)bb * HEADS + hh) * DHEAD;
    const float* Kg = K + ((long)bb * HEADS + hh) * DHEAD;
    const float* Vg = V + ((long)bb * HEADS + hh) * DHEAD;

#pragma unroll
    for (int i = 0; i < (QTILE * DHEAD) / (4 * NTHREADS); ++i) {
        int t4  = i * NTHREADS + tid;
        int row = t4 >> 4;
        int cg  = (t4 & 15) << 2;
        int s   = qb + row;
        float4 f = make_float4(0.f, 0.f, 0.f, 0.f);
        if (s < sq) f = *(const float4*)(Qg + (long)s * rowStride + cg);
        bf16_t* d = &Qs[row * QSTR + cg];
        d[0] = (bf16_t)f.x; d[1] = (bf16_t)f.y; d[2] = (bf16_t)f.z; d[3] = (bf16_t)f.w;
    }
    __syncthreads();

    const int qrow  = wave * 16 + l16;
    const int qbase = qrow * QSTR;
    const v16bf qa0 = pack16(Qs, qbase +      half * 16);
    const v16bf qa1 = pack16(Qs, qbase + 32 + half * 16);

    v8f o[4] = {{}, {}, {}, {}};
    float rmax[8], rsum[8];
#pragma unroll
    for (int v = 0; v < 8; ++v) { rmax[v] = -3.0e38f; rsum[v] = 0.f; }

    int limit = qb + QTILE; if (limit > sq) limit = sq;
    const int ntiles = (limit + KTILE - 1) / KTILE;

    for (int t = 0; t < ntiles; ++t) {
        const int kb = t * KTILE;
        __syncthreads();
#pragma unroll
        for (int i = 0; i < (KTILE * DHEAD) / (4 * NTHREADS); ++i) {
            int t4  = i * NTHREADS + tid;
            int row = t4 >> 4;
            int cg  = (t4 & 15) << 2;
            long src = (long)(kb + row) * rowStride + cg;
            float4 fk = *(const float4*)(Kg + src);
            float4 fv = *(const float4*)(Vg + src);
            bf16_t* dk = &Ks[row * KSTR + cg];
            dk[0] = (bf16_t)fk.x; dk[1] = (bf16_t)fk.y;
            dk[2] = (bf16_t)fk.z; dk[3] = (bf16_t)fk.w;
            Vts[(cg + 0) * VSTR + row] = (bf16_t)fv.x;
            Vts[(cg + 1) * VSTR + row] = (bf16_t)fv.y;
            Vts[(cg + 2) * VSTR + row] = (bf16_t)fv.z;
            Vts[(cg + 3) * VSTR + row] = (bf16_t)fv.w;
        }
        __syncthreads();

        if (t + 1 < ntiles) {
            int prow = tid >> 1;
            int pcol = (tid & 1) << 5;
            if (kb + KTILE + prow < sq) {
                long nsrc = (long)(kb + KTILE + prow) * rowStride + pcol;
                __builtin_prefetch(Kg + nsrc, 0, 3);
                __builtin_prefetch(Vg + nsrc, 0, 3);
            }
        }

        v8f s[4];
#pragma unroll
        for (int g = 0; g < 4; ++g) {
            v16bf b0 = pack16(Ks, (g * 16 + l16) * KSTR +      half * 16);
            v16bf b1 = pack16(Ks, (g * 16 + l16) * KSTR + 32 + half * 16);
            v8f acc = {};
            acc = __builtin_amdgcn_wmma_f32_16x16x32_bf16(false, qa0, false, b0, (short)0, acc, false, false);
            acc = __builtin_amdgcn_wmma_f32_16x16x32_bf16(false, qa1, false, b1, (short)0, acc, false, false);
            s[g] = acc;
        }

        const float scale = 0.125f;
        const bool need_mask = (kb + KTILE) > qb;
        float p[4][8], tmax[8];
#pragma unroll
        for (int v = 0; v < 8; ++v) {
            int rg = qb + wave * 16 + v + 8 * half;
            float m = -3.0e38f;
#pragma unroll
            for (int g = 0; g < 4; ++g) {
                float x = s[g][v] * scale;
                if (need_mask && (kb + g * 16 + l16 > rg)) x = -3.0e38f;
                p[g][v] = x;
                m = fmaxf(m, x);
            }
            tmax[v] = m;
        }
#pragma unroll
        for (int v = 0; v < 8; ++v) tmax[v] = red16_max(tmax[v]);

        float alpha[8], lsum[8];
#pragma unroll
        for (int v = 0; v < 8; ++v) {
            float nm = fmaxf(rmax[v], tmax[v]);
            alpha[v] = __expf(rmax[v] - nm);
            rmax[v]  = nm;
            float acc = 0.f;
#pragma unroll
            for (int g = 0; g < 4; ++g) {
                float e = __expf(p[g][v] - nm);
                p[g][v] = e;
                acc += e;
            }
            lsum[v] = acc;
        }
#pragma unroll
        for (int v = 0; v < 8; ++v) lsum[v] = red16_sum(lsum[v]);
#pragma unroll
        for (int v = 0; v < 8; ++v) {
            rsum[v] = rsum[v] * alpha[v] + lsum[v];
            o[0][v] *= alpha[v]; o[1][v] *= alpha[v];
            o[2][v] *= alpha[v]; o[3][v] *= alpha[v];
        }

        bf16_t* Pw = &Ps[wave * 16 * PSTR];
#pragma unroll
        for (int v = 0; v < 8; ++v) {
            int pr = v + 8 * half;
#pragma unroll
            for (int g = 0; g < 4; ++g)
                Pw[pr * PSTR + g * 16 + l16] = (bf16_t)p[g][v];
        }
        __syncthreads();

        const int pbase = l16 * PSTR;
        v16bf pa0 = pack16(Pw, pbase +      half * 16);
        v16bf pa1 = pack16(Pw, pbase + 32 + half * 16);

#pragma unroll
        for (int j = 0; j < 4; ++j) {
            v16bf vb0 = pack16(Vts, (j * 16 + l16) * VSTR +      half * 16);
            v16bf vb1 = pack16(Vts, (j * 16 + l16) * VSTR + 32 + half * 16);
            o[j] = __builtin_amdgcn_wmma_f32_16x16x32_bf16(false, pa0, false, vb0, (short)0, o[j], false, false);
            o[j] = __builtin_amdgcn_wmma_f32_16x16x32_bf16(false, pa1, false, vb1, (short)0, o[j], false, false);
        }
    }

#pragma unroll
    for (int v = 0; v < 8; ++v) {
        int rg = qb + wave * 16 + v + 8 * half;
        if (rg < sq) {
            float inv = 1.0f / rsum[v];
            long off = ((long)rg * B + bb) * (HEADS * DHEAD) + hh * DHEAD + l16;
            Out[off +  0] = o[0][v] * inv;
            Out[off + 16] = o[1][v] * inv;
            Out[off + 32] = o[2][v] * inv;
            Out[off + 48] = o[3][v] * inv;
        }
    }
}

extern "C" void kernel_launch(void* const* d_in, const int* in_sizes, int n_in,
                              void* d_out, int out_size, void* d_ws, size_t ws_size,
                              hipStream_t stream) {
    const float* Q = (const float*)d_in[0];
    const float* K = (const float*)d_in[1];
    const float* V = (const float*)d_in[2];
    float* Out = (float*)d_out;

    long qn   = (long)in_sizes[0];
    long mask = (long)in_sizes[3];
    long sb   = qn / (HEADS * DHEAD);     // sq * B
    int  sq   = (int)(mask / sb);
    int  B    = (int)(sb / sq);

    dim3 grid((sq + QTILE - 1) / QTILE, B * HEADS);
    size_t n = (size_t)sq * B * HEADS * DHEAD;

    if (ws_size >= 3 * n * sizeof(unsigned short)) {
        bf16_t* Qw = (bf16_t*)d_ws;
        bf16_t* Kw = Qw + n;
        bf16_t* Vw = Kw + n;
        long blocks = (long)((n / 4 + 255) / 256);
        convert_pack_kernel<<<dim3((unsigned)blocks), 256, 0, stream>>>(Q, K, V, Qw, Kw, Vw, sq, B);
        attn_tdm_kernel<<<grid, NTHREADS, 0, stream>>>(Qw, Kw, Vw, Out, sq, B);
    } else {
        attn_fallback_kernel<<<grid, NTHREADS, 0, stream>>>(Q, K, V, Out, sq, B);
    }
}
